// FuncMRegressor_89601607729230
// MI455X (gfx1250) — matrix operand order repaired and checked by reading, not compiled, not run on hardware
//
#include <hip/hip_runtime.h>

// ---------------------------------------------------------------------------
// 2-layer LSTM (units=512, in=256, B=256, T=256) + Dense(1) head for gfx1250.
// bf16 WMMA (v_wmma_f32_16x16x32_bf16) GEMMs, f32 accumulation, fused gates.
// Dual-M waves: each wave computes 32(batch) x 16(unit) x 4 gates = 8 WMMAs
// per K-step, reusing every B fragment twice.
// ---------------------------------------------------------------------------

typedef __bf16 bf16_t;
typedef __attribute__((ext_vector_type(16))) __bf16 v16bf;
typedef __attribute__((ext_vector_type(8)))  __bf16 v8bf;
typedef __attribute__((ext_vector_type(8)))  float  v8f;

#define UNITS  512
#define IN_DIM 256
#define BATCH  256
#define SEQ    256
#define GATES4 (4 * UNITS)   // 2048 gate columns (i,f,g,o blocks of 512)

__device__ __forceinline__ float sigmoid_f(float x) {
    return 1.0f / (1.0f + __expf(-x));
}
__device__ __forceinline__ float tanh_f(float x) {
    float e = __expf(2.0f * x);
    return (e - 1.0f) / (e + 1.0f);
}

// --- fp32 -> bf16 bulk convert (for X) -------------------------------------
__global__ void f32_to_bf16_kernel(const float* __restrict__ src,
                                   bf16_t* __restrict__ dst, int n) {
    int i = blockIdx.x * blockDim.x + threadIdx.x;
    if (i < n) dst[i] = (bf16_t)src[i];
}

// --- build transposed concatenated weights: WT[n*Ktot + k], n in [0,2048) --
// k <  K1 : input-kernel row (W[k, n]) ; k >= K1 : recurrent row (U[k-K1, n])
__global__ void build_wt_kernel(const float* __restrict__ W,
                                const float* __restrict__ U,
                                bf16_t* __restrict__ WT, int K1, int Ktot) {
    int i = blockIdx.x * blockDim.x + threadIdx.x;
    int total = GATES4 * Ktot;
    if (i >= total) return;
    int n = i / Ktot;
    int k = i - n * Ktot;
    float v = (k < K1) ? W[(size_t)k * GATES4 + n]
                       : U[(size_t)(k - K1) * GATES4 + n];
    WT[i] = (bf16_t)v;
}

__global__ void zero_u32_kernel(unsigned int* __restrict__ p, int n) {
    int i = blockIdx.x * blockDim.x + threadIdx.x;
    if (i < n) p[i] = 0u;
}

// --- one LSTM layer for one timestep ---------------------------------------
// z = [A1 | A2] @ WTcat^T + bias ; gates -> c_out, h_out (bf16)
// Wave tile: 32 batch rows (two 16-row WMMA fragments) x 16 units x 4 gates.
__global__ __launch_bounds__(128)
void lstm_step_kernel(const bf16_t* __restrict__ A1, int ldA1, int K1,
                      const bf16_t* __restrict__ A2, int ldA2,
                      const bf16_t* __restrict__ WT, int Ktot,
                      const float* __restrict__ bias,
                      const float* c_in, float* c_out,
                      bf16_t* __restrict__ h_out) {
    const int lane   = threadIdx.x & 31;
    const int wave   = blockIdx.x * (blockDim.x >> 5) + (threadIdx.x >> 5); // 0..255
    const int mpair  = wave >> 5;          // 0..7  : 32-row batch slab
    const int n_tile = wave & 31;          // 0..31 : unit tile
    const int m0 = mpair << 5;             // first batch row of the slab
    const int n0 = n_tile << 4;
    const int hihalf = lane >> 4;          // 0 or 1
    const int lo = lane & 15;
    const int n = n0 + lo;                 // B/C/D column owned by this lane

    const int akb = hihalf << 3;           // A K-base per lane half: 0 / 8
    const int bkb = hihalf << 3;           // B K-base per lane half: 0 / 8

    // A rows for the two M fragments owned by this lane
    const int mA0 = m0 + lo;
    const int mA1 = m0 + 16 + lo;
    const bf16_t* a1r0 = A1 + (size_t)mA0 * ldA1;
    const bf16_t* a1r1 = A1 + (size_t)mA1 * ldA1;
    const bf16_t* a2r0 = A2 + (size_t)mA0 * ldA2;
    const bf16_t* a2r1 = A2 + (size_t)mA1 * ldA2;

    const bf16_t* wtg[4];
#pragma unroll
    for (int g = 0; g < 4; ++g)
        wtg[g] = WT + (size_t)(g * UNITS + n) * Ktot;

    v8f acc[2][4] = {};   // [m-fragment][gate]

    for (int k0 = 0; k0 < Ktot; k0 += 32) {
        // ---- two A fragments: 16x32 bf16, ISA layout (two 8-elem K chunks)
        const bf16_t* ar0 = (k0 < K1) ? (a1r0 + k0) : (a2r0 + (k0 - K1));
        const bf16_t* ar1 = (k0 < K1) ? (a1r1 + k0) : (a2r1 + (k0 - K1));
        union { v16bf v; v8bf h[2]; } a0, a1;
        a0.h[0] = *(const v8bf*)(ar0 + akb);
        a0.h[1] = *(const v8bf*)(ar0 + akb + 16);
        a1.h[0] = *(const v8bf*)(ar1 + akb);
        a1.h[1] = *(const v8bf*)(ar1 + akb + 16);

        // ---- 4 B fragments: per lane two 8-elem K chunks (transposed W), same order as A
        const int kb = k0 + bkb;
        v16bf bfrag[4];
#pragma unroll
        for (int g = 0; g < 4; ++g) {
            union { v16bf v; v8bf h[2]; } bt;
            bt.h[0] = *(const v8bf*)(wtg[g] + kb);
            bt.h[1] = *(const v8bf*)(wtg[g] + kb + 16);
            bfrag[g] = bt.v;
            __builtin_prefetch(wtg[g] + kb + 32, 0, 3);  // next K-slab -> near cache
        }

        // ---- 8 independent WMMAs: every B fragment reused for both M frags
#pragma unroll
        for (int g = 0; g < 4; ++g) {
            acc[0][g] = __builtin_amdgcn_wmma_f32_16x16x32_bf16(
                false, a0.v, false, bfrag[g], (short)0, acc[0][g], false, false);
            acc[1][g] = __builtin_amdgcn_wmma_f32_16x16x32_bf16(
                false, a1.v, false, bfrag[g], (short)0, acc[1][g], false, false);
        }
    }

    // ---- fused gate epilogue (per-lane: N fixed, M varies over 8 VGPRs) ----
    const float bi = bias[0 * UNITS + n];
    const float bf = bias[1 * UNITS + n];
    const float bg = bias[2 * UNITS + n];
    const float bo = bias[3 * UNITS + n];

#pragma unroll
    for (int mt = 0; mt < 2; ++mt) {
#pragma unroll
        for (int j = 0; j < 8; ++j) {
            const int M = m0 + (mt << 4) + j + (hihalf << 3);
            const size_t idx = (size_t)M * UNITS + n;
            const float cprev = c_in[idx];
            const float iv = sigmoid_f(acc[mt][0][j] + bi);
            const float fv = sigmoid_f(acc[mt][1][j] + bf);
            const float gv = tanh_f(acc[mt][2][j] + bg);
            const float ov = sigmoid_f(acc[mt][3][j] + bo);
            const float cn = fv * cprev + iv * gv;
            c_out[idx] = cn;
            h_out[idx] = (bf16_t)(ov * tanh_f(cn));
        }
    }
}

// --- Dense(1) head: out[b] = h1[b,:] . Wfc + bfc ---------------------------
__global__ void head_kernel(const bf16_t* __restrict__ h1,
                            const float* __restrict__ Wfc,
                            const float* __restrict__ bfc,
                            float* __restrict__ out) {
    int b = blockIdx.x * blockDim.x + threadIdx.x;
    if (b >= BATCH) return;
    float s = 0.0f;
    const bf16_t* row = h1 + (size_t)b * UNITS;
#pragma unroll 8
    for (int k = 0; k < UNITS; ++k) s += (float)row[k] * Wfc[k];
    out[b] = s + bfc[0];
}

// ---------------------------------------------------------------------------
extern "C" void kernel_launch(void* const* d_in, const int* in_sizes, int n_in,
                              void* d_out, int out_size, void* d_ws, size_t ws_size,
                              hipStream_t stream) {
    (void)in_sizes; (void)n_in; (void)out_size; (void)ws_size;
    const float* X   = (const float*)d_in[0];   // [B,T,D]
    const float* W0  = (const float*)d_in[1];   // [256,2048]
    const float* U0  = (const float*)d_in[2];   // [512,2048]
    const float* b0  = (const float*)d_in[3];   // [2048]
    const float* W1  = (const float*)d_in[4];   // [512,2048]
    const float* U1  = (const float*)d_in[5];   // [512,2048]
    const float* b1  = (const float*)d_in[6];   // [2048]
    const float* Wfc = (const float*)d_in[7];   // [512]
    const float* bfc = (const float*)d_in[8];   // [1]
    float* out = (float*)d_out;                 // [256]

    char* ws = (char*)d_ws;
    size_t off = 0;
    auto walloc = [&](size_t bytes) -> void* {
        void* p = ws + off;
        off += (bytes + 255) & ~(size_t)255;
        return p;
    };

    const int K0tot = IN_DIM + UNITS;   // 768
    const int K1tot = 2 * UNITS;        // 1024

    bf16_t* Xbf = (bf16_t*)walloc((size_t)BATCH * SEQ * IN_DIM * sizeof(bf16_t));
    bf16_t* WT0 = (bf16_t*)walloc((size_t)GATES4 * K0tot * sizeof(bf16_t));
    bf16_t* WT1 = (bf16_t*)walloc((size_t)GATES4 * K1tot * sizeof(bf16_t));
    bf16_t* H0buf[2] = { (bf16_t*)walloc((size_t)BATCH * UNITS * sizeof(bf16_t)),
                         (bf16_t*)walloc((size_t)BATCH * UNITS * sizeof(bf16_t)) };
    bf16_t* H1buf[2] = { (bf16_t*)walloc((size_t)BATCH * UNITS * sizeof(bf16_t)),
                         (bf16_t*)walloc((size_t)BATCH * UNITS * sizeof(bf16_t)) };
    float* C0 = (float*)walloc((size_t)BATCH * UNITS * sizeof(float));
    float* C1 = (float*)walloc((size_t)BATCH * UNITS * sizeof(float));

    // --- one-time (per call) conversions & state init ---
    {
        int nX = BATCH * SEQ * IN_DIM;
        f32_to_bf16_kernel<<<(nX + 255) / 256, 256, 0, stream>>>(X, Xbf, nX);

        int nW0 = GATES4 * K0tot;
        build_wt_kernel<<<(nW0 + 255) / 256, 256, 0, stream>>>(W0, U0, WT0, IN_DIM, K0tot);
        int nW1 = GATES4 * K1tot;
        build_wt_kernel<<<(nW1 + 255) / 256, 256, 0, stream>>>(W1, U1, WT1, UNITS, K1tot);

        int nHu32 = (BATCH * UNITS * (int)sizeof(bf16_t)) / 4;
        int nCu32 = (BATCH * UNITS * (int)sizeof(float)) / 4;
        zero_u32_kernel<<<(nHu32 + 255) / 256, 256, 0, stream>>>((unsigned*)H0buf[0], nHu32);
        zero_u32_kernel<<<(nHu32 + 255) / 256, 256, 0, stream>>>((unsigned*)H0buf[1], nHu32);
        zero_u32_kernel<<<(nHu32 + 255) / 256, 256, 0, stream>>>((unsigned*)H1buf[0], nHu32);
        zero_u32_kernel<<<(nHu32 + 255) / 256, 256, 0, stream>>>((unsigned*)H1buf[1], nHu32);
        zero_u32_kernel<<<(nCu32 + 255) / 256, 256, 0, stream>>>((unsigned*)C0, nCu32);
        zero_u32_kernel<<<(nCu32 + 255) / 256, 256, 0, stream>>>((unsigned*)C1, nCu32);
    }

    // --- sequential recurrence: 2 dependent step kernels per timestep ---
    // 256 dual-M waves per layer-step -> 64 blocks x 4 waves (wave32).
    const dim3 grid(64), block(128);
    for (int t = 0; t < SEQ; ++t) {
        const bf16_t* h0r = H0buf[t & 1];
        bf16_t*       h0w = H0buf[(t + 1) & 1];
        const bf16_t* h1r = H1buf[t & 1];
        bf16_t*       h1w = H1buf[(t + 1) & 1];

        // layer 0: A = [x_t | h0_prev], K = 256 + 512
        lstm_step_kernel<<<grid, block, 0, stream>>>(
            Xbf + (size_t)t * IN_DIM, SEQ * IN_DIM, IN_DIM,
            h0r, UNITS, WT0, K0tot, b0, C0, C0, h0w);

        // layer 1: A = [h0_new | h1_prev], K = 512 + 512
        lstm_step_kernel<<<grid, block, 0, stream>>>(
            h0w, UNITS, UNITS,
            h1r, UNITS, WT1, K1tot, b1, C1, C1, h1w);
    }

    // final h1 lives in H1buf[SEQ & 1] == H1buf[0]
    head_kernel<<<1, 256, 0, stream>>>(H1buf[SEQ & 1], Wfc, bfc, out);
}
